// Quantizer2d_15547781611765
// MI455X (gfx1250) — matrix operand order repaired and checked by reading, not compiled, not run on hardware
//
#include <hip/hip_runtime.h>
#include <hip/hip_bf16.h>

// ---------------------------------------------------------------------------
// VQ quantizer forward for MI455X (gfx1250, wave32, WMMA, async-to-LDS).
// x: [32, 512, 32, 32] fp32, weight: [1024, 512] fp32.
// outputs (flat): q [32,512,32,32] fp32, codebook_loss, commitment_loss,
//                 indices [32,32,32] (as float).
// argmin_k ||t - w_k||^2 == argmin_k ( ||w_k||^2 - 2 t.w_k )  -> GEMM + argmin.
// GEMM kernel: A panel (64 tok x 512 K bf16) LDS-resident (async staged once);
// B panels (64 codes x 512 K) double-buffered via global_load_async_to_lds_b128.
// Each wave computes TWO 16x16 C tiles sharing one A operand: 6 ds_load_b128
// feed 2 back-to-back v_wmma_f32_16x16x32_bf16 per K step.
// ---------------------------------------------------------------------------

typedef __attribute__((ext_vector_type(16))) __bf16 v16bf;
typedef __attribute__((ext_vector_type(8)))  __bf16 v8bf;
typedef __attribute__((ext_vector_type(4)))  __bf16 v4bf;
typedef __attribute__((ext_vector_type(8)))  float  v8f;

#define NB       32          // batch
#define NC       512         // channels / token dim
#define NP       1024        // H*W tokens per batch image
#define NT       (NB * NP)   // 32768 tokens total
#define NK       1024        // codebook size
#define Q_ELEMS  (NT * NC)   // 16777216

#define TOK_PER_BLK 64       // tokens per block in GEMM kernel
#define NCHUNK      64       // codes scored per B panel (2 subtiles / wave)
#define KSTEP       32       // K depth per WMMA (bf16 16x16x32)
#define NKS         (NC / KSTEP)   // 16 K steps per panel
#define PITCH       (NC + 8) // LDS row pitch (260 dwords -> 4-bank row skew)

// async copy of one 16-byte chunk, global -> LDS (ASYNCcnt-tracked)
#define ASYNC_CP16(ldsoff, gptr)                                      \
  asm volatile("global_load_async_to_lds_b128 %0, %1, off"            \
               :: "v"(ldsoff), "v"(gptr) : "memory")

#define S_WAIT_ASYNC(imm)                                             \
  asm volatile("s_wait_asynccnt " #imm ::: "memory")

// ---------------------------------------------------------------------------
// Kernel 1: wsq[k] = sum_c w^2 ; wbf = bf16(weight) ; zero loss accumulator.
// ---------------------------------------------------------------------------
__global__ void k_wsq(const float* __restrict__ weight,
                      float* __restrict__ wsq,
                      __bf16* __restrict__ wbf,
                      float* __restrict__ loss_acc) {
  __shared__ float red[256];
  const int k   = blockIdx.x;
  const int tid = threadIdx.x;
  float s = 0.f;
  for (int c = tid; c < NC; c += 256) {
    float w = weight[(size_t)k * NC + c];
    wbf[(size_t)k * NC + c] = (__bf16)w;
    s += w * w;
  }
  red[tid] = s;
  __syncthreads();
  for (int off = 128; off > 0; off >>= 1) {
    if (tid < off) red[tid] += red[tid + off];
    __syncthreads();
  }
  if (tid == 0) {
    wsq[k] = red[0];
    if (k == 0) *loss_acc = 0.f;   // stream order serializes before k_gather
  }
}

// ---------------------------------------------------------------------------
// Kernel 2: x[b][c][p] fp32 -> xbf[b*NP+p][c] bf16 via 32x32 LDS tile.
// ---------------------------------------------------------------------------
__global__ void k_xcast(const float* __restrict__ x,
                        __bf16* __restrict__ xbf) {
  __shared__ float tile[32][33];           // [c][p], padded
  const int blk = blockIdx.x;              // 32 b * 16 ct * 32 pt
  const int b   = blk >> 9;
  const int ct  = (blk >> 5) & 15;
  const int pt  = blk & 31;
  const int tid = threadIdx.x;

  const int col  = tid & 31;
  const int row0 = tid >> 5;               // 0..7
#pragma unroll
  for (int i = 0; i < 4; ++i) {
    const int c = ct * 32 + row0 + 8 * i;
    tile[row0 + 8 * i][col] =
        x[((size_t)(b * NC + c)) * NP + pt * 32 + col];
  }
  __syncthreads();

  const int pr = tid >> 3;                 // 0..31
  const int cg = (tid & 7) * 4;            // 0,4,...,28
  v4bf o;
#pragma unroll
  for (int j = 0; j < 4; ++j) o[j] = (__bf16)tile[cg + j][pr];
  __bf16* dst =
      xbf + ((size_t)(b * NP + pt * 32 + pr)) * NC + ct * 32 + cg;
  *(v4bf*)dst = o;
}

// ---------------------------------------------------------------------------
// Kernel 3: scores + argmin via WMMA bf16 with async-to-LDS staging.
// Block: 256 threads = 8 waves arranged 4 (M) x 2 (N); each wave owns two
// 16-code subtiles -> block scores 64 tokens x 64 codes per panel.
// ---------------------------------------------------------------------------
__global__ void k_gemm_argmin(const __bf16* __restrict__ xbf,
                              const __bf16* __restrict__ wbf,
                              const float* __restrict__ wsq,
                              int*   __restrict__ idxbuf,
                              float* __restrict__ out_idx) {
  __shared__ __align__(64) __bf16 Abuf[TOK_PER_BLK][PITCH];   //  ~65 KB
  __shared__ __align__(64) __bf16 Bbuf[2][NCHUNK][PITCH];     // ~130 KB
  __shared__ float redv[4][TOK_PER_BLK];
  __shared__ int   redi[4][TOK_PER_BLK];

  const int tid  = threadIdx.x;
  const int wave = tid >> 5;
  const int lane = tid & 31;
  const int lh   = lane >> 4;     // lane half (K split per ISA layout)
  const int l15  = lane & 15;
  const int mi   = wave >> 1;     // 0..3 : 16-token row tile
  const int ni   = wave & 1;      // 0..1 : 16-code col tile pair

  const int tokbase = blockIdx.x * TOK_PER_BLK;

  // ---- stage resident A panel: 64 x 512 bf16 = 4096 x 16B, 16/thread ----
#pragma unroll
  for (int i = 0; i < 16; ++i) {
    const int q   = tid + 256 * i;
    const int row = q >> 6;
    const int cik = q & 63;
    const __bf16* g = xbf + (size_t)(tokbase + row) * NC + cik * 8;
    ASYNC_CP16((unsigned)(size_t)&Abuf[row][cik * 8], g);
  }
  // ---- stage B panel 0: 64 x 512 bf16 = 4096 x 16B, 16/thread ----
#pragma unroll
  for (int i = 0; i < 16; ++i) {
    const int q   = tid + 256 * i;
    const int row = q >> 6;
    const int cik = q & 63;
    const __bf16* g = wbf + (size_t)row * NC + cik * 8;
    ASYNC_CP16((unsigned)(size_t)&Bbuf[0][row][cik * 8], g);
  }

  float minv0[8], minv1[8];
  int   mini0[8], mini1[8];
#pragma unroll
  for (int r = 0; r < 8; ++r) {
    minv0[r] = 3.4e38f; mini0[r] = 0;
    minv1[r] = 3.4e38f; mini1[r] = 0;
  }

  const int m   = mi * 16 + l15;         // A row for this lane
  const int nl0 = ni * 16 + l15;         // B row, subtile 0 (codes 0..31)
  const int nl1 = 32 + nl0;              // B row, subtile 1 (codes 32..63)

  for (int nc = 0; nc < NK / NCHUNK; ++nc) {   // 16 panels
    const int n0  = nc * NCHUNK;
    const int cur = nc & 1;

    if (nc + 1 < NK / NCHUNK) {
      // prefetch next B panel into the other buffer (WAR-safe: barrier at
      // the end of the previous iteration ordered all readers before us)
#pragma unroll
      for (int i = 0; i < 16; ++i) {
        const int q   = tid + 256 * i;
        const int row = q >> 6;
        const int cik = q & 63;
        const __bf16* g = wbf + (size_t)(n0 + NCHUNK + row) * NC + cik * 8;
        ASYNC_CP16((unsigned)(size_t)&Bbuf[cur ^ 1][row][cik * 8], g);
      }
      S_WAIT_ASYNC(0x10);  // async loads retire in order: current panel done
    } else {
      S_WAIT_ASYNC(0x0);
    }
    __syncthreads();       // panels visible to all waves

    const __bf16* arow  = &Abuf[m][0];
    const __bf16* brow0 = &Bbuf[cur][nl0][0];
    const __bf16* brow1 = &Bbuf[cur][nl1][0];

    v8f acc0 = {}, acc1 = {};
#pragma unroll
    for (int ks = 0; ks < NKS; ++ks) {          // 16 K steps, 32 WMMAs
      const int kb = ks * KSTEP;
      // A operand: lanes0-15 K {0..7,16..23}, lanes16-31 K {8..15,24..31}
      v8bf alo = *(const v8bf*)(arow + kb + lh * 8);
      v8bf ahi = *(const v8bf*)(arow + kb + 16 + lh * 8);
      v16bf av = __builtin_shufflevector(alo, ahi, 0, 1, 2, 3, 4, 5, 6, 7,
                                         8, 9, 10, 11, 12, 13, 14, 15);
      // B operands: lanes0-15 K {0..7,16..23}, lanes16-31 K {8..15,24..31}
      v8bf b0l = *(const v8bf*)(brow0 + kb + lh * 8);
      v8bf b0h = *(const v8bf*)(brow0 + kb + 16 + lh * 8);
      v16bf bv0 = __builtin_shufflevector(b0l, b0h, 0, 1, 2, 3, 4, 5, 6, 7,
                                          8, 9, 10, 11, 12, 13, 14, 15);
      v8bf b1l = *(const v8bf*)(brow1 + kb + lh * 8);
      v8bf b1h = *(const v8bf*)(brow1 + kb + 16 + lh * 8);
      v16bf bv1 = __builtin_shufflevector(b1l, b1h, 0, 1, 2, 3, 4, 5, 6, 7,
                                          8, 9, 10, 11, 12, 13, 14, 15);
      // two independent accumulator chains sharing the A operand
      acc0 = __builtin_amdgcn_wmma_f32_16x16x32_bf16(
          false, av, false, bv0, (short)0, acc0, false, false);
      acc1 = __builtin_amdgcn_wmma_f32_16x16x32_bf16(
          false, av, false, bv1, (short)0, acc1, false, false);
    }

    // distances: d = ||w||^2 - 2 * dot  (two code columns per lane)
    const int ng0 = n0 + nl0;
    const int ng1 = n0 + nl1;
    const float w20 = wsq[ng0];
    const float w21 = wsq[ng1];
#pragma unroll
    for (int r = 0; r < 8; ++r) {
      const float s0 = w20 - 2.0f * acc0[r];
      if (s0 < minv0[r] || (s0 == minv0[r] && ng0 < mini0[r])) {
        minv0[r] = s0; mini0[r] = ng0;
      }
      const float s1 = w21 - 2.0f * acc1[r];
      if (s1 < minv1[r] || (s1 == minv1[r] && ng1 < mini1[r])) {
        minv1[r] = s1; mini1[r] = ng1;
      }
    }
    __syncthreads();       // readers done before next overwrite
  }

  // reduce across the 16 N-columns held by lanes of the same half
  for (int off = 1; off < 16; off <<= 1) {
#pragma unroll
    for (int r = 0; r < 8; ++r) {
      float ov = __shfl_xor(minv0[r], off, 32);
      int   oi = __shfl_xor(mini0[r], off, 32);
      if (ov < minv0[r] || (ov == minv0[r] && oi < mini0[r])) {
        minv0[r] = ov; mini0[r] = oi;
      }
      ov = __shfl_xor(minv1[r], off, 32);
      oi = __shfl_xor(mini1[r], off, 32);
      if (ov < minv1[r] || (ov == minv1[r] && oi < mini1[r])) {
        minv1[r] = ov; mini1[r] = oi;
      }
    }
  }

  // lanes 0 and 16 hold per-token results; merge 4 column groups via LDS.
  if (l15 == 0) {
#pragma unroll
    for (int r = 0; r < 8; ++r) {
      const int m_local = mi * 16 + lh * 8 + r;
      redv[ni * 2 + 0][m_local] = minv0[r];
      redi[ni * 2 + 0][m_local] = mini0[r];
      redv[ni * 2 + 1][m_local] = minv1[r];
      redi[ni * 2 + 1][m_local] = mini1[r];
    }
  }
  __syncthreads();
  if (tid < TOK_PER_BLK) {
    float bv = redv[0][tid];
    int   bi = redi[0][tid];
#pragma unroll
    for (int g = 1; g < 4; ++g) {
      const float v = redv[g][tid];
      const int   i = redi[g][tid];
      if (v < bv || (v == bv && i < bi)) { bv = v; bi = i; }
    }
    const int t = tokbase + tid;
    idxbuf[t]  = bi;
    out_idx[t] = (float)bi;
  }
}

// ---------------------------------------------------------------------------
// Kernel 4: q = gathered codebook rows (fp32), accumulate (quant - x)^2.
// ---------------------------------------------------------------------------
__global__ void k_gather_loss(const float* __restrict__ x,
                              const float* __restrict__ weight,
                              const int* __restrict__ idxbuf,
                              float* __restrict__ q,
                              float* __restrict__ loss_acc) {
  __shared__ float red[256];
  const int tid    = threadIdx.x;
  const int dp     = tid & 31;     // token within chunk (coalesced dim)
  const int c0     = tid >> 5;     // 0..7
  const int tchunk = blockIdx.x * 32;
  const int b      = tchunk / NP;
  const int pInB   = (tchunk % NP) + dp;

  const int idx = idxbuf[tchunk + dp];
  const float* wrow = weight + (size_t)idx * NC;

  float s = 0.f;
  for (int c = c0; c < NC; c += 8) {
    const size_t a = ((size_t)(b * NC + c)) * NP + pInB;
    const float wv = wrow[c];
    const float xv = x[a];
    q[a] = wv;                      // straight-through: q == quant
    const float d = wv - xv;
    s += d * d;
  }
  red[tid] = s;
  __syncthreads();
  for (int off = 128; off > 0; off >>= 1) {
    if (tid < off) red[tid] += red[tid + off];
    __syncthreads();
  }
  if (tid == 0) atomicAdd(loss_acc, red[0]);
}

// ---------------------------------------------------------------------------
// Kernel 5: finalize the two (numerically identical) loss scalars.
// ---------------------------------------------------------------------------
__global__ void k_final(const float* __restrict__ loss_acc,
                        float* __restrict__ out) {
  const float mean = *loss_acc * (1.0f / (float)Q_ELEMS);
  out[Q_ELEMS]     = mean;  // codebook_loss
  out[Q_ELEMS + 1] = mean;  // commitment_loss
}

// ---------------------------------------------------------------------------
extern "C" void kernel_launch(void* const* d_in, const int* in_sizes, int n_in,
                              void* d_out, int out_size, void* d_ws, size_t ws_size,
                              hipStream_t stream) {
  const float* x      = (const float*)d_in[0];   // [32,512,32,32]
  const float* weight = (const float*)d_in[1];   // [1024,512]
  float* out = (float*)d_out;

  // workspace layout (≈33.3 MB total)
  char* ws = (char*)d_ws;
  float*  wsq      = (float*)(ws);                   //   4 KB
  float*  loss_acc = (float*)(ws + 4096);            //   4 B (padded)
  int*    idxbuf   = (int*)(ws + 8192);              // 128 KB
  __bf16* wbf      = (__bf16*)(ws + 8192 + 131072);  //   1 MB
  __bf16* xbf      = (__bf16*)(ws + 8192 + 131072 + (size_t)NK * NC * 2); // 32 MB

  float* out_q   = out;                    // 16777216
  float* out_idx = out + Q_ELEMS + 2;      // 32768 (as float)

  k_wsq<<<NK, 256, 0, stream>>>(weight, wsq, wbf, loss_acc);
  k_xcast<<<NB * (NC / 32) * (NP / 32), 256, 0, stream>>>(x, xbf);
  k_gemm_argmin<<<NT / TOK_PER_BLK, 256, 0, stream>>>(xbf, wbf, wsq,
                                                      idxbuf, out_idx);
  k_gather_loss<<<NT / 32, 256, 0, stream>>>(x, weight, idxbuf, out_q,
                                             loss_acc);
  k_final<<<1, 1, 0, stream>>>(loss_acc, out);
}
